// PPYOLOEHead_90056874262891
// MI455X (gfx1250) — compile-verified, matrix operand order checked by reading, not run on hardware
//
#include <hip/hip_runtime.h>
#include <hip/hip_bf16.h>

#define EPS_F 1e-9f

typedef __attribute__((ext_vector_type(16))) _Float16 v16h;
typedef __attribute__((ext_vector_type(8)))  float    v8f;

// ---------------------------------------------------------------------------
// K1: pairwise IoU + inside-flag.  grid=(ceil(L/256), n, B)
// ---------------------------------------------------------------------------
__global__ void __launch_bounds__(256)
k1_iou(const float* __restrict__ gt_bboxes,   // [B,n,4]
       const float* __restrict__ pred_bboxes, // [B,L,4]
       const float* __restrict__ anchor_pts,  // [L,2]
       float* __restrict__ ws_iou,            // [B,n,L]
       unsigned char* __restrict__ ws_inside, // [B,n,L]
       int n, int L)
{
    int b = blockIdx.z, g = blockIdx.y;
    int l = blockIdx.x * 256 + threadIdx.x;
    if (l >= L) return;
    const float* gb = gt_bboxes + ((size_t)(b * n + g)) * 4;
    float gx0 = gb[0], gy0 = gb[1], gx1 = gb[2], gy1 = gb[3];
    float4 p = ((const float4*)pred_bboxes)[(size_t)b * L + l];
    float ix0 = fmaxf(gx0, p.x), iy0 = fmaxf(gy0, p.y);
    float ix1 = fminf(gx1, p.z), iy1 = fminf(gy1, p.w);
    float iw = fmaxf(ix1 - ix0, 0.f), ih = fmaxf(iy1 - iy0, 0.f);
    float inter = iw * ih;
    float aa = (gx1 - gx0) * (gy1 - gy0);
    float ab = (p.z - p.x) * (p.w - p.y);
    float iou = inter / (aa + ab - inter + EPS_F);
    size_t idx = ((size_t)(b * n + g)) * L + l;
    ws_iou[idx] = iou;
    float2 ap = ((const float2*)anchor_pts)[l];
    float dmin = fminf(fminf(ap.x - gx0, ap.y - gy0), fminf(gx1 - ap.x, gy1 - ap.y));
    ws_inside[idx] = (dmin > EPS_F) ? (unsigned char)1 : (unsigned char)0;
}

// ---------------------------------------------------------------------------
// K2: per (b,g) row: async-stage iou row to LDS, build masked metric in place,
// iterative top-k with lowest-index tie-break, write mask_positive bytes into
// the TRANSPOSED mask [B,L,n].  grid=(n,B), block=256.
// ---------------------------------------------------------------------------
__global__ void __launch_bounds__(256)
k2_topk(const float* __restrict__ ws_iou,
        const unsigned char* __restrict__ ws_inside,
        const float* __restrict__ pred_scores, // [B,L,C]
        const int*   __restrict__ gt_labels,   // [B,n]
        const float* __restrict__ pad_gt,      // [B,n]
        unsigned char* __restrict__ ws_mask,   // [B,L,n] (pre-zeroed)
        const int* __restrict__ topk_ptr,
        int n, int L, int C)
{
    __shared__ float sv[8448];
    __shared__ float rv[256];
    __shared__ int   ri[256];

    int b = blockIdx.y, g = blockIdx.x;
    int tk = topk_ptr[0];
    int cls = gt_labels[b * n + g];
    float padv = pad_gt[b * n + g];
    size_t rowoff = ((size_t)(b * n + g)) * L;

    // ---- async copy of the raw IoU row (global -> LDS), ASYNCcnt-tracked ----
    for (int l = threadIdx.x; l < L; l += 256) {
        unsigned lds_addr = (unsigned)(size_t)(&sv[l]);
        unsigned long long ga = (unsigned long long)(size_t)(ws_iou + rowoff + l);
        asm volatile("global_load_async_to_lds_b32 %0, %1, off"
                     :: "v"(lds_addr), "v"(ga) : "memory");
    }
    asm volatile("s_wait_asynccnt 0x0" ::: "memory");
    __syncthreads();

    // ---- masked alignment metric in place: inside * score * iou^6 ----
    for (int l = threadIdx.x; l < L; l += 256) {
        float v  = sv[l];
        float v2 = v * v;
        float v6 = v2 * v2 * v2;
        __builtin_prefetch(&pred_scores[((size_t)b * L + l + 256) * C + cls], 0, 1);
        float s  = pred_scores[((size_t)b * L + l) * C + cls];
        sv[l] = (ws_inside[rowoff + l] ? 1.f : 0.f) * s * v6;
    }
    __syncthreads();

    for (int k = 0; k < tk; k++) {
        float best = -2.f; int bi = 0x7fffffff;
        for (int l = threadIdx.x; l < L; l += 256) {
            float v = sv[l];
            if (v > best) { best = v; bi = l; }   // lowest index wins ties
        }
        rv[threadIdx.x] = best; ri[threadIdx.x] = bi;
        __syncthreads();
        for (int s = 128; s > 0; s >>= 1) {
            if (threadIdx.x < s) {
                float ov = rv[threadIdx.x + s]; int oi = ri[threadIdx.x + s];
                if (ov > rv[threadIdx.x] ||
                    (ov == rv[threadIdx.x] && oi < ri[threadIdx.x])) {
                    rv[threadIdx.x] = ov; ri[threadIdx.x] = oi;
                }
            }
            __syncthreads();
        }
        if (threadIdx.x == 0) {
            int w = ri[0];
            // mask_positive = is_in_topk * is_in_gts * pad  (transposed layout)
            ws_mask[((size_t)b * L + w) * n + g] =
                (ws_inside[rowoff + w] && padv > 0.f) ? (unsigned char)1
                                                      : (unsigned char)0;
            sv[w] = -1.f;                         // exclude from next round
        }
        __syncthreads();
    }
}

// ---------------------------------------------------------------------------
// K3: mask_positive_sum via WMMA: counts[b,l] = sum_g mask[b,l,g].
// Each wave handles 16 anchors; A = 16x32 f16 0/1 tile (documented layout:
// lane%16 = anchor row, halfs h -> K = h+(h>=8?8:0)+(lane>=16?8:0)),
// B = ones, so D[:,0] holds exact integer counts.  Branchless packed build:
// two u64 loads per step from the contiguous [B,L,n] mask row.
// block=256 (8 waves), grid = ceil(B*(L/16)/8)
// ---------------------------------------------------------------------------
__global__ void __launch_bounds__(256)
k3_colsum_wmma(const unsigned char* __restrict__ ws_mask, // [B,L,n]
               float* __restrict__ ws_colsum,             // [B,L]
               int n, int L, int ntiles)
{
    int wave = threadIdx.x >> 5;
    int lane = threadIdx.x & 31;
    int tile = blockIdx.x * 8 + wave;
    if (tile >= ntiles) return;                  // wave-uniform branch
    int tpb = L >> 4;
    int b   = tile / tpb;
    int a0  = (tile - b * tpb) << 4;
    int l   = a0 + (lane & 15);
    int khalf = (lane >> 4) ? 8 : 0;

    const unsigned char* mrow = ws_mask + ((size_t)b * L + l) * n;

    v16h ones;
#pragma unroll
    for (int h = 0; h < 16; h++) ones[h] = (_Float16)1.0f;
    v8f c = {};

    if (n == 128) {
#pragma unroll
        for (int kb = 0; kb < 128; kb += 32) {
            unsigned long long m0 = *(const unsigned long long*)(mrow + kb + khalf);
            unsigned long long m1 = *(const unsigned long long*)(mrow + kb + khalf + 16);
            v16h a;
#pragma unroll
            for (int h = 0; h < 8; h++)
                a[h] = (_Float16)(float)((m0 >> (8 * h)) & 1ull);
#pragma unroll
            for (int h = 0; h < 8; h++)
                a[h + 8] = (_Float16)(float)((m1 >> (8 * h)) & 1ull);
            c = __builtin_amdgcn_wmma_f32_16x16x32_f16(false, a, false, ones,
                                                       (short)0, c, false, false);
        }
    } else {
        for (int kb = 0; kb < n; kb += 32) {
            unsigned long long m0 = *(const unsigned long long*)(mrow + kb + khalf);
            unsigned long long m1 = *(const unsigned long long*)(mrow + kb + khalf + 16);
            v16h a;
#pragma unroll
            for (int h = 0; h < 8; h++)
                a[h] = (_Float16)(float)((m0 >> (8 * h)) & 1ull);
#pragma unroll
            for (int h = 0; h < 8; h++)
                a[h + 8] = (_Float16)(float)((m1 >> (8 * h)) & 1ull);
            c = __builtin_amdgcn_wmma_f32_16x16x32_f16(false, a, false, ones,
                                                       (short)0, c, false, false);
        }
    }
    // C/D layout: lanes 0/16 hold column N=0; VGPR r -> rows M=r / M=8+r.
    if ((lane & 15) == 0) {
        int mbase = (lane >= 16) ? 8 : 0;
#pragma unroll
        for (int r = 0; r < 8; r++)
            ws_colsum[(size_t)b * L + a0 + mbase + r] = c[r];
    }
}

// ---------------------------------------------------------------------------
// K4: per-anchor conflict resolution + labels/bboxes + per-GT atomic maxes.
// ---------------------------------------------------------------------------
__global__ void __launch_bounds__(256)
k4_resolve(const unsigned char* __restrict__ ws_mask,   // [B,L,n]
           const float* __restrict__ ws_iou,            // [B,n,L]
           const float* __restrict__ ws_colsum,
           const int*   __restrict__ gt_labels,
           const float* __restrict__ pred_scores,
           const float* __restrict__ gt_bboxes,
           int* __restrict__ ws_g, int* __restrict__ ws_lab,
           float* __restrict__ ws_m,
           unsigned int* __restrict__ max_m, unsigned int* __restrict__ max_i,
           float* __restrict__ out_labels, float* __restrict__ out_bboxes,
           int B, int n, int L, int C)
{
    int idx = blockIdx.x * 256 + threadIdx.x;
    if (idx >= B * L) return;
    int b = idx / L, l = idx - b * L;
    float cnt = ws_colsum[idx];
    int g = -1;
    if (cnt > 1.5f) {                            // multiple -> argmax IoU (first max)
        size_t base = ((size_t)b * n) * L + l;
        float best = -1.f;
        for (int gg = 0; gg < n; gg++) {
            float v = ws_iou[base + (size_t)gg * L];
            if (v > best) { best = v; g = gg; }
        }
    } else if (cnt > 0.5f) {                     // single -> first set byte
        const unsigned long long* mr =
            (const unsigned long long*)(ws_mask + (size_t)idx * n);
        for (int w = 0; w < (n >> 3); w++) {
            unsigned long long mv = mr[w];
            if (mv) { g = (w << 3) + (__builtin_ctzll(mv) >> 3); break; }
        }
    }
    bool positive = (g >= 0);
    int gi  = positive ? g : 0;
    int gl  = gt_labels[b * n + gi];
    int lab = positive ? gl : C;
    float m = 0.f;
    if (positive) {
        float iouv = ws_iou[((size_t)b * n + gi) * L + l];
        float v2 = iouv * iouv;
        float s  = pred_scores[((size_t)b * L + l) * C + gl];
        m = s * v2 * v2 * v2;
        atomicMax(&max_m[b * n + gi], __float_as_uint(m));    // non-neg f32 bit order
        atomicMax(&max_i[b * n + gi], __float_as_uint(iouv));
    }
    ws_g[idx] = gi; ws_lab[idx] = lab; ws_m[idx] = m;
    out_labels[idx] = (float)lab;
    float4 gb = ((const float4*)gt_bboxes)[b * n + gi];
    ((float4*)out_bboxes)[idx] = gb;
}

// ---------------------------------------------------------------------------
// K5: one-hot rescaled scores [B,L,C].
// ---------------------------------------------------------------------------
__global__ void __launch_bounds__(256)
k5_scores(const int* __restrict__ ws_g, const int* __restrict__ ws_lab,
          const float* __restrict__ ws_m,
          const unsigned int* __restrict__ max_m, const unsigned int* __restrict__ max_i,
          float* __restrict__ out_scores,
          int n, int L, int C, long long total)
{
    long long t = (long long)blockIdx.x * 256 + threadIdx.x;
    if (t >= total) return;
    int c = (int)(t % C);
    long long al = t / C;                        // b*L + l
    int b = (int)(al / L);
    int lab = ws_lab[al];
    float v = 0.f;
    if (lab < C && c == lab) {
        int g = ws_g[al];
        float mm = __uint_as_float(max_m[b * n + g]);
        float mi = __uint_as_float(max_i[b * n + g]);
        v = ws_m[al] / (mm + EPS_F) * mi;
    }
    out_scores[t] = v;
}

// ---------------------------------------------------------------------------
static inline size_t align16(size_t x) { return (x + 15) & ~(size_t)15; }

extern "C" void kernel_launch(void* const* d_in, const int* in_sizes, int n_in,
                              void* d_out, int out_size, void* d_ws, size_t ws_size,
                              hipStream_t stream) {
    const float* pred_scores  = (const float*)d_in[0];  // [B,L,C]
    const float* pred_bboxes  = (const float*)d_in[1];  // [B,L,4]
    const float* anchor_pts   = (const float*)d_in[2];  // [L,2]
    const int*   gt_labels    = (const int*)  d_in[3];  // [B,n]
    const float* gt_bboxes    = (const float*)d_in[4];  // [B,n,4]
    const float* pad_gt       = (const float*)d_in[5];  // [B,n,1]
    const int*   topk_ptr     = (const int*)  d_in[7];  // device scalar

    const int L = in_sizes[2] / 2;
    const int B = in_sizes[1] / (L * 4);
    const int n = in_sizes[3] / B;
    const int C = in_sizes[0] / (B * L);

    const size_t BnL = (size_t)B * n * L;
    const size_t BL  = (size_t)B * L;

    size_t off = 0;
    float* ws_iou = (float*)((char*)d_ws + off);                    off += align16(BnL * 4);
    unsigned char* ws_inside = (unsigned char*)((char*)d_ws + off); off += align16(BnL);
    unsigned char* ws_mask   = (unsigned char*)((char*)d_ws + off); off += align16(BnL);
    float* ws_colsum = (float*)((char*)d_ws + off);                 off += align16(BL * 4);
    int*   ws_g      = (int*)  ((char*)d_ws + off);                 off += align16(BL * 4);
    int*   ws_lab    = (int*)  ((char*)d_ws + off);                 off += align16(BL * 4);
    float* ws_m      = (float*)((char*)d_ws + off);                 off += align16(BL * 4);
    unsigned int* max_m = (unsigned int*)((char*)d_ws + off);       off += align16((size_t)B * n * 4);
    unsigned int* max_i = (unsigned int*)((char*)d_ws + off);       off += align16((size_t)B * n * 4);

    float* out_labels = (float*)d_out;          // [B,L]
    float* out_bboxes = out_labels + BL;        // [B,L,4]
    float* out_scores = out_bboxes + BL * 4;    // [B,L,C]

    hipMemsetAsync(ws_mask, 0, BnL, stream);
    hipMemsetAsync(max_m, 0, (size_t)B * n * 4, stream);
    hipMemsetAsync(max_i, 0, (size_t)B * n * 4, stream);

    dim3 g1((L + 255) / 256, n, B);
    k1_iou<<<g1, 256, 0, stream>>>(gt_bboxes, pred_bboxes, anchor_pts,
                                   ws_iou, ws_inside, n, L);

    dim3 g2(n, B);
    k2_topk<<<g2, 256, 0, stream>>>(ws_iou, ws_inside, pred_scores, gt_labels,
                                    pad_gt, ws_mask, topk_ptr, n, L, C);

    int ntiles = B * (L / 16);
    k3_colsum_wmma<<<(ntiles + 7) / 8, 256, 0, stream>>>(ws_mask, ws_colsum,
                                                         n, L, ntiles);

    k4_resolve<<<(int)((BL + 255) / 256), 256, 0, stream>>>(
        ws_mask, ws_iou, ws_colsum, gt_labels, pred_scores, gt_bboxes,
        ws_g, ws_lab, ws_m, max_m, max_i, out_labels, out_bboxes, B, n, L, C);

    long long total = (long long)BL * C;
    k5_scores<<<(int)((total + 255) / 256), 256, 0, stream>>>(
        ws_g, ws_lab, ws_m, max_m, max_i, out_scores, n, L, C, total);
}
